// KMISPooling_44736379355628
// MI455X (gfx1250) — compile-verified, matrix operand order checked by reading, not run on hardware
//
#include <hip/hip_runtime.h>
#include <hip/hip_bf16.h>

// ---------------------------------------------------------------------------
// KMIS pooling pipeline for gfx1250 (MI455X).
//   score GEMV -> TDM (tensor_load_to_lds) -> LDS -> V_WMMA_F32_16X16X4_F32
//   graph hops -> atomic min/max/add streams (memory bound)
//   rank       -> deterministic bucket sort (65536 MSD buckets)
// ---------------------------------------------------------------------------

typedef float v2f  __attribute__((ext_vector_type(2)));
typedef float v8f  __attribute__((ext_vector_type(8)));
typedef _Float16 v16h __attribute__((ext_vector_type(16)));
typedef int  v4i  __attribute__((ext_vector_type(4)));
typedef int  v8i  __attribute__((ext_vector_type(8)));

#define NBUCK 65536
#define HCAP  (1u << 22)
#define HMASK (HCAP - 1u)
#define HEMPTY 0xFFFFFFFFFFFFFFFFull
#define KMIS_ITERS 24

// ---------------------------- utility fills --------------------------------
__global__ void fill_i32_kernel(int* p, int v, int cnt) {
    int i = blockIdx.x * blockDim.x + threadIdx.x;
    if (i < cnt) p[i] = v;
}
__global__ void fill_f32_kernel(float* p, float v, int cnt) {
    int i = blockIdx.x * blockDim.x + threadIdx.x;
    if (i < cnt) p[i] = v;
}
__global__ void fill_u64_kernel(unsigned long long* p, unsigned long long v, int cnt) {
    int i = blockIdx.x * blockDim.x + threadIdx.x;
    if (i < cnt) p[i] = v;
}
__global__ void copy_f32_kernel(float* dst, const float* src, int cnt) {
    int i = blockIdx.x * blockDim.x + threadIdx.x;
    if (i < cnt) dst[i] = src[i];
}
__global__ void copy_i32_kernel(int* dst, const int* src, int cnt) {
    int i = blockIdx.x * blockDim.x + threadIdx.x;
    if (i < cnt) dst[i] = src[i];
}

// ------------------------- score GEMV via TDM + WMMA -----------------------
// One wave (32 threads) computes 16 rows of sigmoid(x @ w + b).
// x tile (16 x d) and w are DMA'd into LDS by the Tensor Data Mover, with a
// 1-dword pad every 256 dwords so tile rows sit at pitch 257 (bank spread).
// NOTE: &sh[0] is passed to the asm as a dummy operand so the LDS allocation
// escapes -- otherwise alias analysis assumes the asm cannot write it and
// folds all LDS reads to undef (observed in round 2 asm).
#if __has_builtin(__builtin_amdgcn_wmma_f32_16x16x4_f32)
__global__ __launch_bounds__(32) void score_wmma_kernel(
    const float* __restrict__ x, const float* __restrict__ w,
    const float* __restrict__ b, float* __restrict__ score, int n, int d) {
    __shared__ float sh[16 * 257 + 256];     // x tile (padded) + w staging
    const int lane = threadIdx.x;            // 0..31
    const int rowBase = blockIdx.x * 16;
    const int m = lane & 15;                 // row this lane feeds into A
    const int hi = lane >> 4;                // 0 or 1
    const int kHalf = hi * 2;                // lanes 0-15: K 0,1 ; 16-31: K 2,3
    const bool pad = (d == 256);
    const int pitch = d + (pad ? 1 : 0);

    // ---- Tensor Data Mover: global -> LDS (async, TENSORcnt tracked) ----
    {
        unsigned long long gx = (unsigned long long)(uintptr_t)(x + (size_t)rowBase * d);
        int remRows = n - rowBase;           // rows left; TDM zero-fills past edge
        v4i g0;
        g0[0] = 1;                                               // count=1 (valid)
        g0[1] = 0;                                               // lds_addr = 0
        g0[2] = (int)(unsigned)(gx & 0xffffffffull);
        g0[3] = (int)((unsigned)((gx >> 32) & 0x1ffffffull) | (2u << 30));  // type=2
        v8i g1;
        unsigned w0 = 0x00020000u;                               // data_size = 4B
        if (pad) w0 |= (1u << 20) | (7u << 22);                  // pad 1 dw / 256 dw
        g1[0] = (int)w0;
        g1[1] = (int)(((unsigned)d & 0xffffu) << 16);            // tensor_dim0 lo16
        g1[2] = (int)((((unsigned)d >> 16) & 0xffffu)
                      | (((unsigned)remRows & 0xffffu) << 16));  // dim0 hi / dim1 lo
        g1[3] = (int)((((unsigned)remRows >> 16) & 0xffffu)
                      | (((unsigned)d & 0xffffu) << 16));        // dim1 hi / tile_dim0
        g1[4] = 16;                                              // tile_dim1 = 16 rows
        g1[5] = d;                                               // dim0 stride lo32
        g1[6] = (int)(((unsigned)d & 0xffffu) << 16);            // dim1 stride lo16
        g1[7] = 0;

        unsigned long long gw = (unsigned long long)(uintptr_t)w;
        v4i h0;
        h0[0] = 1;
        h0[1] = (int)(16u * (unsigned)pitch * 4u);               // w after x tile
        h0[2] = (int)(unsigned)(gw & 0xffffffffull);
        h0[3] = (int)((unsigned)((gw >> 32) & 0x1ffffffull) | (2u << 30));
        v8i h1;
        h1[0] = 0x00020000;                                      // 4B, no pad
        h1[1] = (int)(((unsigned)d & 0xffffu) << 16);
        h1[2] = (int)((((unsigned)d >> 16) & 0xffffu) | (1u << 16));  // tensor_dim1=1
        h1[3] = (int)(((unsigned)d & 0xffffu) << 16);            // tile_dim0 = d
        h1[4] = 1;                                               // tile_dim1 = 1
        h1[5] = d;
        h1[6] = 0;
        h1[7] = 0;

        float* shp = &sh[0];   // escape the LDS buffer into the asm
        asm volatile("tensor_load_to_lds %0, %1\n\t"
                     "tensor_load_to_lds %2, %3\n\t"
                     "s_wait_tensorcnt 0x0"
                     :: "s"(g0), "s"(g1), "s"(h0), "s"(h1), "v"(shp)
                     : "memory");
    }
    __builtin_amdgcn_s_wait_tensorcnt(0);

    const float* ax = &sh[m * pitch];
    const float* wl = &sh[16 * pitch];
    const float nmask = ((lane & 15) == 0) ? 1.0f : 0.0f;  // branchless B column 0

    v8f acc = {};
    for (int k0 = 0; k0 < d; k0 += 4) {
        v2f a;
        a.x = ax[k0 + kHalf];
        a.y = ax[k0 + kHalf + 1];
        v2f bm;
        bm.x = nmask * wl[k0 + hi];
        bm.y = nmask * wl[k0 + 2 + hi];
        acc = __builtin_amdgcn_wmma_f32_16x16x4_f32(
            false, a, false, bm, (short)0, acc, false, false);
    }
    // Column N=0 of D: lane 0 holds M=0..7 in acc[0..7], lane 16 holds M=8..15.
    if ((lane & 15) == 0) {
        int mOff = hi * 8;
        for (int r = 0; r < 8; ++r) {
            int orow = rowBase + mOff + r;
            if (orow < n) {
                float s = acc[r] + b[0];
                score[orow] = 1.0f / (1.0f + __expf(-s));
            }
        }
    }
}
#else
// Fallback: f16 WMMA 16x16x32 (codegen-confirmed builtin).
__global__ __launch_bounds__(32) void score_wmma_kernel(
    const float* __restrict__ x, const float* __restrict__ w,
    const float* __restrict__ b, float* __restrict__ score, int n, int d) {
    const int lane = threadIdx.x;
    const int rowBase = blockIdx.x * 16;
    const int m = lane & 15;
    const int kb = (lane >> 4) * 8;
    int row = rowBase + m;
    if (row >= n) row = n - 1;
    const float* xr = x + (size_t)row * d;
    v8f acc = {};
    for (int k0 = 0; k0 < d; k0 += 32) {
        v16h a, bm;
        for (int j = 0; j < 8; ++j) {
            a[j]     = (_Float16)xr[k0 + kb + j];
            a[j + 8] = (_Float16)xr[k0 + 16 + kb + j];
        }
        for (int j = 0; j < 16; ++j) bm[j] = (_Float16)0.f;
        if ((lane & 15) == 0) {
            for (int j = 0; j < 16; ++j) {
                int kk = k0 + 2 * j + (lane >> 4);
                if (kk < d) bm[j] = (_Float16)w[kk];
            }
        }
        acc = __builtin_amdgcn_wmma_f32_16x16x32_f16(
            false, a, false, bm, (short)0, acc, false, false);
    }
    if ((lane & 15) == 0) {
        int mOff = (lane >> 4) * 8;
        for (int r = 0; r < 8; ++r) {
            int orow = rowBase + mOff + r;
            if (orow < n) {
                float s = acc[r] + b[0];
                score[orow] = 1.0f / (1.0f + __expf(-s));
            }
        }
    }
}
#endif

// ------------------------------ ks hops ------------------------------------
__global__ void edge_ksum_kernel(const int* __restrict__ row, const int* __restrict__ col,
                                 const float* __restrict__ u, float* __restrict__ v, int e) {
    int i = blockIdx.x * blockDim.x + threadIdx.x;
    if (i >= e) return;
    __builtin_prefetch(&row[min(i + 8192, e - 1)], 0, 1);
    __builtin_prefetch(&col[min(i + 8192, e - 1)], 0, 1);
    atomicAdd(&v[col[i]], u[row[i]]);
}

// --------------------------- key + histogram -------------------------------
__device__ __forceinline__ unsigned int order_key(float f) {
    unsigned int u = __float_as_uint(f);
    return (u & 0x80000000u) ? ~u : (u | 0x80000000u);  // ascending == float ascending
}
__global__ void key_hist_kernel(const float* __restrict__ score, const float* __restrict__ ks,
                                unsigned int* __restrict__ key, unsigned int* __restrict__ hist,
                                int n) {
    int i = blockIdx.x * blockDim.x + threadIdx.x;
    if (i >= n) return;
    float upd = score[i] / ks[i];
    unsigned int k = order_key(-upd);   // ascending key == descending upd
    key[i] = k;
    atomicAdd(&hist[k >> 16], 1u);
}

__global__ __launch_bounds__(1024) void scan_hist_kernel(const unsigned int* __restrict__ hist,
                                                         int* __restrict__ bucketStart) {
    __shared__ int part[1024];
    int t = threadIdx.x;
    int base = t * 64;
    int sum = 0;
    for (int i = 0; i < 64; ++i) sum += (int)hist[base + i];
    part[t] = sum;
    __syncthreads();
    for (int off = 1; off < 1024; off <<= 1) {
        int v = (t >= off) ? part[t - off] : 0;
        __syncthreads();
        part[t] += v;
        __syncthreads();
    }
    int excl = (t == 0) ? 0 : part[t - 1];
    for (int i = 0; i < 64; ++i) {
        bucketStart[base + i] = excl;
        excl += (int)hist[base + i];
    }
}

__global__ void bucket_scatter_kernel(const unsigned int* __restrict__ key,
                                      const int* __restrict__ bucketStart,
                                      unsigned int* __restrict__ bucketCnt,
                                      int* __restrict__ slot, int n) {
    int i = blockIdx.x * blockDim.x + threadIdx.x;
    if (i >= n) return;
    unsigned int bkt = key[i] >> 16;
    int pos = bucketStart[bkt] + (int)atomicAdd(&bucketCnt[bkt], 1u);
    slot[pos] = i;
}

// Deterministic rank: within-bucket ties broken by (full key, index).
__global__ void rank_kernel(const unsigned int* __restrict__ key,
                            const int* __restrict__ bucketStart,
                            const unsigned int* __restrict__ hist,
                            const int* __restrict__ slot,
                            int* __restrict__ rank, int n) {
    int i = blockIdx.x * blockDim.x + threadIdx.x;
    if (i >= n) return;
    unsigned int ki = key[i];
    unsigned int bkt = ki >> 16;
    int start = bucketStart[bkt];
    int cnt = (int)hist[bkt];
    int r = 0;
    for (int j = 0; j < cnt; ++j) {
        int o = slot[start + j];
        unsigned int ko = key[o];
        r += (ko < ki) || (ko == ki && o < i);
    }
    rank[i] = start + r;
}

// ------------------------------ k-MIS loop ---------------------------------
__global__ void edge_min_kernel(const int* __restrict__ row, const int* __restrict__ col,
                                const int* __restrict__ mr, int* __restrict__ tmp, int e) {
    int i = blockIdx.x * blockDim.x + threadIdx.x;
    if (i >= e) return;
    __builtin_prefetch(&row[min(i + 8192, e - 1)], 0, 1);
    atomicMin(&tmp[col[i]], mr[row[i]]);
}
__global__ void combine_min_kernel(int* __restrict__ mr, int* __restrict__ tmp, int n) {
    int i = blockIdx.x * blockDim.x + threadIdx.x;
    if (i >= n) return;
    int m = mr[i], t = tmp[i];
    mr[i] = m < t ? m : t;
    tmp[i] = 0x7fffffff;
}
__global__ void mis_update_kernel(const int* __restrict__ rank, const int* __restrict__ mr,
                                  int* __restrict__ mis, int* __restrict__ mcur,
                                  int* __restrict__ tmp, int n) {
    int i = blockIdx.x * blockDim.x + threadIdx.x;
    if (i >= n) return;
    int s = mis[i] | (rank[i] == mr[i]);
    mis[i] = s;
    mcur[i] = s;
    tmp[i] = 0;
}
__global__ void edge_max_kernel(const int* __restrict__ row, const int* __restrict__ col,
                                const int* __restrict__ m, int* __restrict__ tmp, int e) {
    int i = blockIdx.x * blockDim.x + threadIdx.x;
    if (i >= e) return;
    __builtin_prefetch(&col[min(i + 8192, e - 1)], 0, 1);
    atomicMax(&tmp[col[i]], m[row[i]]);
}
__global__ void combine_max_kernel(int* __restrict__ m, int* __restrict__ tmp, int n) {
    int i = blockIdx.x * blockDim.x + threadIdx.x;
    if (i >= n) return;
    int t = tmp[i];
    t = t > 1 ? 1 : t;
    int v = m[i];
    m[i] = v > t ? v : t;
    tmp[i] = 0;
}
__global__ void iter_final_kernel(const int* __restrict__ mcur, const int* __restrict__ rank,
                                  int* __restrict__ mr, int* __restrict__ tmp, int n) {
    int i = blockIdx.x * blockDim.x + threadIdx.x;
    if (i >= n) return;
    mr[i] = (mcur[i] > 0) ? n : rank[i];
    tmp[i] = 0x7fffffff;
}
__global__ void mr_from_mis_kernel(const int* __restrict__ mis, const int* __restrict__ rank,
                                   int* __restrict__ mr, int* __restrict__ tmp, int n) {
    int i = blockIdx.x * blockDim.x + threadIdx.x;
    if (i >= n) return;
    mr[i] = mis[i] ? rank[i] : n;
    tmp[i] = 0x7fffffff;
}

// --------------------------- MIS compaction --------------------------------
__global__ __launch_bounds__(256) void blkcount_kernel(const int* __restrict__ mis,
                                                       int* __restrict__ blockCnt, int n) {
    __shared__ int sh[256];
    int t = threadIdx.x;
    int i = blockIdx.x * 256 + t;
    sh[t] = (i < n) ? (mis[i] ? 1 : 0) : 0;
    __syncthreads();
    for (int off = 128; off > 0; off >>= 1) {
        if (t < off) sh[t] += sh[t + off];
        __syncthreads();
    }
    if (t == 0) blockCnt[blockIdx.x] = sh[0];
}
__global__ __launch_bounds__(1024) void scan_blocks_kernel(const int* __restrict__ blockCnt,
                                                           int* __restrict__ blockOff,
                                                           int* __restrict__ counters, int nB) {
    __shared__ int sh[1024];
    int t = threadIdx.x;
    int v = (t < nB) ? blockCnt[t] : 0;
    sh[t] = v;
    __syncthreads();
    for (int off = 1; off < 1024; off <<= 1) {
        int u = (t >= off) ? sh[t - off] : 0;
        __syncthreads();
        sh[t] += u;
        __syncthreads();
    }
    if (t < nB) blockOff[t] = sh[t] - v;
    if (t == 1023) counters[0] = sh[1023];   // total MIS count
}
__global__ __launch_bounds__(256) void scatter_mis_kernel(const int* __restrict__ mis,
                                                          const int* __restrict__ rank,
                                                          const int* __restrict__ blockOff,
                                                          int* __restrict__ mis_idx,
                                                          int* __restrict__ rank2cluster, int n) {
    __shared__ int sh[256];
    int t = threadIdx.x;
    int i = blockIdx.x * 256 + t;
    int f = (i < n) ? (mis[i] ? 1 : 0) : 0;
    sh[t] = f;
    __syncthreads();
    for (int off = 1; off < 256; off <<= 1) {
        int u = (t >= off) ? sh[t - off] : 0;
        __syncthreads();
        sh[t] += u;
        __syncthreads();
    }
    if (f) {
        int pos = blockOff[blockIdx.x] + sh[t] - 1;
        mis_idx[pos] = i;
        rank2cluster[rank[i]] = pos;
    }
}
__global__ void cluster_kernel(const int* __restrict__ mr, const int* __restrict__ rank2cluster,
                               int* __restrict__ cluster, int n) {
    int i = blockIdx.x * blockDim.x + threadIdx.x;
    if (i >= n) return;
    int m = mr[i];
    cluster[i] = (m < n) ? rank2cluster[m] : 0;
}

// ------------------------ pair unique via hashing --------------------------
__device__ __forceinline__ unsigned int hash_u64(unsigned long long k) {
    unsigned long long h = k * 0x9E3779B97F4A7C15ull;
    h ^= h >> 31;
    return (unsigned int)h & HMASK;
}
__global__ void hash_insert_kernel(const int* __restrict__ row, const int* __restrict__ col,
                                   const int* __restrict__ cluster,
                                   unsigned long long* __restrict__ hkey,
                                   unsigned int* __restrict__ hcnt, int e) {
    int i = blockIdx.x * blockDim.x + threadIdx.x;
    if (i >= e) return;
    unsigned long long c0 = (unsigned int)cluster[row[i]];
    unsigned long long c1 = (unsigned int)cluster[col[i]];
    unsigned long long k = (c0 << 32) | c1;
    unsigned int s = hash_u64(k);
    for (;;) {
        unsigned long long old = atomicCAS(&hkey[s], HEMPTY, k);
        if (old == HEMPTY || old == k) { atomicAdd(&hcnt[s], 1u); break; }
        s = (s + 1u) & HMASK;
    }
}
__global__ void hash_count_kernel(const unsigned long long* __restrict__ hkey,
                                  int* __restrict__ keepCnt, int cap) {
    int i = blockIdx.x * blockDim.x + threadIdx.x;
    if (i >= cap) return;
    unsigned long long k = hkey[i];
    if (k == HEMPTY) return;
    unsigned int c0 = (unsigned int)(k >> 32), c1 = (unsigned int)k;
    if (c0 != c1) atomicAdd(keepCnt, 1);
}
__global__ void hash_emit_kernel(const unsigned long long* __restrict__ hkey,
                                 const unsigned int* __restrict__ hcnt,
                                 const int* __restrict__ counters,
                                 int* __restrict__ emitPos,
                                 float* __restrict__ out, int d, int cap) {
    int i = blockIdx.x * blockDim.x + threadIdx.x;
    if (i >= cap) return;
    unsigned long long k = hkey[i];
    if (k == HEMPTY) return;
    unsigned int c0 = (unsigned int)(k >> 32), c1 = (unsigned int)k;
    if (c0 == c1) return;
    long long keep = counters[1];
    long long base = (long long)counters[0] * d;   // edges follow the x_pool block
    int idx = atomicAdd(emitPos, 1);
    int* outI = (int*)out;
    outI[base + idx] = (int)c0;                    // new_edge_index row 0
    outI[base + keep + idx] = (int)c1;             // new_edge_index row 1
    out[base + 2 * keep + idx] = (float)hcnt[i];   // new_edge_attr
}

// ------------------------------- x_pool ------------------------------------
__global__ __launch_bounds__(64) void xpool_kernel(const float* __restrict__ x,
                                                   const float* __restrict__ score,
                                                   const int* __restrict__ mis_idx,
                                                   const int* __restrict__ counters,
                                                   float* __restrict__ out, int d) {
    int j = blockIdx.x;
    if (j >= counters[0]) return;
    int i = mis_idx[j];
    float s = score[i];
    const float* xr = x + (size_t)i * d;
    float* orow = out + (size_t)j * d;
    if ((d & 3) == 0) {
        const float4* x4 = (const float4*)xr;
        float4* o4 = (float4*)orow;
        int d4 = d >> 2;
        for (int c = threadIdx.x; c < d4; c += blockDim.x) {
            float4 v = x4[c];
            v.x *= s; v.y *= s; v.z *= s; v.w *= s;
            o4[c] = v;
        }
    } else {
        for (int c = threadIdx.x; c < d; c += blockDim.x) orow[c] = xr[c] * s;
    }
}

// ------------------------------- launch ------------------------------------
extern "C" void kernel_launch(void* const* d_in, const int* in_sizes, int n_in,
                              void* d_out, int out_size, void* d_ws, size_t ws_size,
                              hipStream_t stream) {
    const float* x  = (const float*)d_in[0];
    const int*   ei = (const int*)d_in[1];
    const float* lw = (const float*)d_in[2];
    const float* lb = (const float*)d_in[3];

    const int d = in_sizes[2];            // 256
    const int n = in_sizes[0] / d;        // 100000
    const int e = in_sizes[1] / 2;        // 1600000
    const int* row = ei;
    const int* col = ei + e;

    // ---- workspace layout ----
    char* ws = (char*)d_ws;
    size_t off = 0;
    auto take = [&](size_t bytes) { char* p = ws + off; off = (off + bytes + 255) & ~(size_t)255; return p; };
    float* score        = (float*)take((size_t)n * 4);
    float* ksU          = (float*)take((size_t)n * 4);
    float* ksV          = (float*)take((size_t)n * 4);
    unsigned int* key   = (unsigned int*)take((size_t)n * 4);
    int* rank           = (int*)take((size_t)n * 4);
    int* mr             = (int*)take((size_t)n * 4);
    int* tmp            = (int*)take((size_t)n * 4);
    int* mis            = (int*)take((size_t)n * 4);
    int* mcur           = (int*)take((size_t)n * 4);
    unsigned int* hist  = (unsigned int*)take((size_t)NBUCK * 4);
    int* bucketStart    = (int*)take((size_t)NBUCK * 4);
    unsigned int* bCnt  = (unsigned int*)take((size_t)NBUCK * 4);
    int* slot           = (int*)take((size_t)n * 4);
    int* blockCnt       = (int*)take(4096 * 4);
    int* blockOff       = (int*)take(4096 * 4);
    int* mis_idx        = (int*)take((size_t)n * 4);
    int* rank2cluster   = (int*)take((size_t)n * 4);
    int* cluster        = (int*)take((size_t)n * 4);
    int* counters       = (int*)take(64 * 4);
    unsigned long long* hkey = (unsigned long long*)take((size_t)HCAP * 8);
    unsigned int* hcnt  = (unsigned int*)take((size_t)HCAP * 4);

    const int TB = 256;
    const int gN = (n + TB - 1) / TB;
    const int gE = (e + TB - 1) / TB;
    const int gH = (NBUCK + TB - 1) / TB;
    const int gC = (HCAP + TB - 1) / TB;
    const int nB = gN;  // 256-sized blocks over n

    // ---- init ----
    fill_i32_kernel<<<1, 64, 0, stream>>>(counters, 0, 64);
    fill_f32_kernel<<<gN, TB, 0, stream>>>(ksU, 1.0f, n);
    fill_f32_kernel<<<gN, TB, 0, stream>>>(ksV, 1.0f, n);
    fill_i32_kernel<<<gH, TB, 0, stream>>>((int*)hist, 0, NBUCK);
    fill_i32_kernel<<<gH, TB, 0, stream>>>((int*)bCnt, 0, NBUCK);
    fill_i32_kernel<<<gN, TB, 0, stream>>>(rank2cluster, 0, n);
    fill_u64_kernel<<<gC, TB, 0, stream>>>(hkey, HEMPTY, HCAP);
    fill_i32_kernel<<<gC, TB, 0, stream>>>((int*)hcnt, 0, HCAP);

    // ---- score = sigmoid(x @ w + b): TDM -> LDS -> WMMA ----
    score_wmma_kernel<<<(n + 15) / 16, 32, 0, stream>>>(x, lw, lb, score, n, d);

    // ---- ks: 2 hops of ks = ks + segsum(ks[row] -> col) ----
    edge_ksum_kernel<<<gE, TB, 0, stream>>>(row, col, ksU, ksV, e);
    copy_f32_kernel<<<gN, TB, 0, stream>>>(ksU, ksV, n);
    edge_ksum_kernel<<<gE, TB, 0, stream>>>(row, col, ksU, ksV, e);

    // ---- deterministic rank of -upd ----
    key_hist_kernel<<<gN, TB, 0, stream>>>(score, ksV, key, hist, n);
    scan_hist_kernel<<<1, 1024, 0, stream>>>(hist, bucketStart);
    bucket_scatter_kernel<<<gN, TB, 0, stream>>>(key, bucketStart, bCnt, slot, n);
    rank_kernel<<<gN, TB, 0, stream>>>(key, bucketStart, hist, slot, rank, n);

    // ---- k-MIS fixed-point loop (idempotent after convergence) ----
    copy_i32_kernel<<<gN, TB, 0, stream>>>(mr, rank, n);
    fill_i32_kernel<<<gN, TB, 0, stream>>>(mis, 0, n);
    fill_i32_kernel<<<gN, TB, 0, stream>>>(tmp, 0x7fffffff, n);
    for (int it = 0; it < KMIS_ITERS; ++it) {
        edge_min_kernel<<<gE, TB, 0, stream>>>(row, col, mr, tmp, e);
        combine_min_kernel<<<gN, TB, 0, stream>>>(mr, tmp, n);
        edge_min_kernel<<<gE, TB, 0, stream>>>(row, col, mr, tmp, e);
        combine_min_kernel<<<gN, TB, 0, stream>>>(mr, tmp, n);
        mis_update_kernel<<<gN, TB, 0, stream>>>(rank, mr, mis, mcur, tmp, n);
        edge_max_kernel<<<gE, TB, 0, stream>>>(row, col, mcur, tmp, e);
        combine_max_kernel<<<gN, TB, 0, stream>>>(mcur, tmp, n);
        edge_max_kernel<<<gE, TB, 0, stream>>>(row, col, mcur, tmp, e);
        combine_max_kernel<<<gN, TB, 0, stream>>>(mcur, tmp, n);
        iter_final_kernel<<<gN, TB, 0, stream>>>(mcur, rank, mr, tmp, n);
    }

    // ---- final mr = khop_min(mis ? rank : n) ----
    mr_from_mis_kernel<<<gN, TB, 0, stream>>>(mis, rank, mr, tmp, n);
    edge_min_kernel<<<gE, TB, 0, stream>>>(row, col, mr, tmp, e);
    combine_min_kernel<<<gN, TB, 0, stream>>>(mr, tmp, n);
    edge_min_kernel<<<gE, TB, 0, stream>>>(row, col, mr, tmp, e);
    combine_min_kernel<<<gN, TB, 0, stream>>>(mr, tmp, n);

    // ---- compact MIS (deterministic scans) ----
    blkcount_kernel<<<nB, TB, 0, stream>>>(mis, blockCnt, n);
    scan_blocks_kernel<<<1, 1024, 0, stream>>>(blockCnt, blockOff, counters, nB);
    scatter_mis_kernel<<<nB, TB, 0, stream>>>(mis, rank, blockOff, mis_idx, rank2cluster, n);
    cluster_kernel<<<gN, TB, 0, stream>>>(mr, rank2cluster, cluster, n);

    // ---- unique pairs with counts ----
    hash_insert_kernel<<<gE, TB, 0, stream>>>(row, col, cluster, hkey, hcnt, e);
    hash_count_kernel<<<gC, TB, 0, stream>>>(hkey, &counters[1], HCAP);
    hash_emit_kernel<<<gC, TB, 0, stream>>>(hkey, hcnt, counters, &counters[2],
                                            (float*)d_out, d, HCAP);

    // ---- x_pool ----
    xpool_kernel<<<n, 64, 0, stream>>>(x, score, mis_idx, counters, (float*)d_out, d);
}
